// ContextualTokenSAE_68616397521272
// MI455X (gfx1250) — matrix operand order repaired and checked by reading, not compiled, not run on hardware
//
#include <hip/hip_runtime.h>
#include <hip/hip_bf16.h>
#include <math.h>

// ---------------------------------------------------------------------------
// ContextualTokenSAE forward for MI455X (gfx1250, wave32, WMMA)
//   z       = x @ W_enc + b_enc                  (16384x384)(384x4096)
//   p       = softmax(z, -1)
//   mask    = adaptive top-k (CDF of sorted p >= 0.9)
//   latents = relu(z) * mask
//   recon   = tanh(latents @ W_dec + b_dec)      (16384x4096)(4096x128)
//
// Roofline: ~69 GFLOP of GEMM vs ~1.3 GB HBM traffic -> bandwidth bound
// (~56 us at 23.3 TB/s). GEMMs use v_wmma_f32_16x16x32_bf16 with fp32->bf16
// LDS staging. The top-k threshold is found with a 4-level radix select on
// the float bit pattern (monotone for positive p) instead of a bitonic sort:
// 4 histogram passes over LDS instead of 78 sort passes.
// ---------------------------------------------------------------------------

typedef __bf16 v16bf __attribute__((ext_vector_type(16)));
typedef __bf16 v8bf  __attribute__((ext_vector_type(8)));
typedef float  v8f   __attribute__((ext_vector_type(8)));

#define S_ROWS   16384
#define D_IN     384
#define D_LAT    4096
#define D_OUT    128
#define TAU      0.9f

// ===========================================================================
// Kernel 1: z = x @ W_enc + b_enc  (writes into latents region as scratch z)
// Block: 256 threads (8 waves). Tile: 128(M) x 128(N), K-step 32.
// Wave grid 4(M) x 2(N); each wave owns 2x4 = eight 16x16 WMMA tiles.
// ===========================================================================
__global__ __launch_bounds__(256) void sae_gemm1_kernel(
    const float* __restrict__ x, const float* __restrict__ Wenc,
    const float* __restrict__ benc, float* __restrict__ z)
{
    __shared__ __align__(128) __bf16 Alds[128 * 32];   // [m][k]
    __shared__ __align__(128) __bf16 Blds[128 * 32];   // [n][k] (transposed)

    const int tid  = threadIdx.x;
    const int row0 = blockIdx.x * 128;
    const int n0   = blockIdx.y * 128;
    const int wave = tid >> 5;
    const int lane = tid & 31;
    const int wm   = wave & 3;        // 0..3 -> M offset wm*32
    const int wn   = wave >> 2;       // 0..1 -> N offset wn*64
    const int l16  = lane & 15;
    const int lh   = lane >> 4;       // lane half (0/1)

    v8f acc[2][4];
#pragma unroll
    for (int a = 0; a < 2; ++a)
#pragma unroll
        for (int b = 0; b < 4; ++b) acc[a][b] = (v8f)(0.0f);

    for (int ks = 0; ks < D_IN; ks += 32) {
        // ---- stage A tile: 128x32 fp32 -> bf16, row-major [m][k] ----
#pragma unroll
        for (int i = 0; i < 4; ++i) {
            const int e  = i * 1024 + tid * 4;
            const int r  = e >> 5;
            const int kk = e & 31;
            const float4 v = *(const float4*)(x + (size_t)(row0 + r) * D_IN + ks + kk);
            Alds[r * 32 + kk + 0] = (__bf16)v.x;
            Alds[r * 32 + kk + 1] = (__bf16)v.y;
            Alds[r * 32 + kk + 2] = (__bf16)v.z;
            Alds[r * 32 + kk + 3] = (__bf16)v.w;
        }
        // ---- stage B tile: 32x128 fp32 -> bf16, transposed to [n][k] ----
#pragma unroll
        for (int i = 0; i < 4; ++i) {
            const int e  = i * 1024 + tid * 4;
            const int kr = e >> 7;
            const int nn = e & 127;
            const float4 v = *(const float4*)(Wenc + (size_t)(ks + kr) * D_LAT + n0 + nn);
            Blds[(nn + 0) * 32 + kr] = (__bf16)v.x;
            Blds[(nn + 1) * 32 + kr] = (__bf16)v.y;
            Blds[(nn + 2) * 32 + kr] = (__bf16)v.z;
            Blds[(nn + 3) * 32 + kr] = (__bf16)v.w;
        }
        // prefetch next k-step (global_prefetch_b8)
        if (ks + 32 < D_IN) {
            __builtin_prefetch(x + (size_t)(row0 + (tid >> 1)) * D_IN + ks + 32, 0, 1);
            __builtin_prefetch(Wenc + (size_t)(ks + 32 + (tid >> 5)) * D_LAT + n0 + (tid & 31) * 4, 0, 1);
        }
        __syncthreads();

        // ---- build fragments: contiguous k = 16*half + s in both A and B ----
        v16bf afrag[2], bfrag[4];
#pragma unroll
        for (int tm = 0; tm < 2; ++tm) {
            const int r  = wm * 32 + tm * 16 + l16;
            const int kb = lh * 16;                      // lanes 0-15: K0..15; 16-31: K16..31
            const v8bf a0 = *(const v8bf*)(&Alds[r * 32 + kb]);
            const v8bf a1 = *(const v8bf*)(&Alds[r * 32 + kb + 8]);
#pragma unroll
            for (int q = 0; q < 8; ++q) { afrag[tm][q] = a0[q]; afrag[tm][q + 8] = a1[q]; }
        }
#pragma unroll
        for (int tn = 0; tn < 4; ++tn) {
            const int c = wn * 64 + tn * 16 + l16;       // lanes 0-15: K0..15; 16-31: K16..31
            bfrag[tn] = *(const v16bf*)(&Blds[c * 32 + lh * 16]);
        }
#pragma unroll
        for (int tm = 0; tm < 2; ++tm)
#pragma unroll
            for (int tn = 0; tn < 4; ++tn)
                acc[tm][tn] = __builtin_amdgcn_wmma_f32_16x16x32_bf16(
                    false, afrag[tm], false, bfrag[tn], (short)0, acc[tm][tn], false, false);
        __syncthreads();
    }

    // ---- epilogue: +bias, store z ----
#pragma unroll
    for (int tm = 0; tm < 2; ++tm) {
#pragma unroll
        for (int tn = 0; tn < 4; ++tn) {
            const int col = n0 + wn * 64 + tn * 16 + l16;
            const float bias = benc[col];
            const int m0 = row0 + wm * 32 + tm * 16 + lh * 8;  // C/D layout: VGPR r -> M=r (+8 for hi lanes)
#pragma unroll
            for (int r = 0; r < 8; ++r)
                z[(size_t)(m0 + r) * D_LAT + col] = acc[tm][tn][r] + bias;
        }
    }
}

// ===========================================================================
// Kernel 2: per-row softmax + adaptive top-k via 4-level float-bit radix
// select; latents = relu(z) * (p >= threshold). One 256-thread block per row.
// The z region is read fully into LDS, then overwritten with the latents.
// All p > 0, so the IEEE bit pattern is monotone in value: we histogram
// probability *sums* into 256 bins per 8-bit digit (MSB first), suffix-scan,
// find the bin where the descending CDF crosses the residual target, and
// descend with the residual. After 4 levels the bin path IS the threshold's
// exact 32-bit pattern.
// ===========================================================================
__global__ __launch_bounds__(256) void sae_topk_kernel(
    const float* zin, float* pout, float* latout)
{
    __shared__ float zrow[D_LAT];
    __shared__ float prow[D_LAT];
    __shared__ float hist[256];
    __shared__ float red[256];
    __shared__ int   schosen;
    __shared__ float starget;

    const int row = blockIdx.x;
    const int tid = threadIdx.x;

    // ---- load z (b128) + row max ----
    const float4* zr4 = (const float4*)(zin + (size_t)row * D_LAT);
    float lmax = -3.402823466e38f;
#pragma unroll
    for (int i = tid; i < D_LAT / 4; i += 256) {
        const float4 v = zr4[i];
        *(float4*)&zrow[i * 4] = v;
        lmax = fmaxf(lmax, fmaxf(fmaxf(v.x, v.y), fmaxf(v.z, v.w)));
    }
    red[tid] = lmax;
    __syncthreads();
#pragma unroll
    for (int s = 128; s > 0; s >>= 1) {
        if (tid < s) red[tid] = fmaxf(red[tid], red[tid + s]);
        __syncthreads();
    }
    const float m = red[0];
    __syncthreads();

    // ---- exp + sum ----
    float lsum = 0.0f;
#pragma unroll
    for (int i = tid; i < D_LAT / 4; i += 256) {
        const float4 v = *(const float4*)&zrow[i * 4];
        float4 e;
        e.x = __expf(v.x - m); e.y = __expf(v.y - m);
        e.z = __expf(v.z - m); e.w = __expf(v.w - m);
        *(float4*)&prow[i * 4] = e;
        lsum += e.x + e.y + e.z + e.w;
    }
    red[tid] = lsum;
    __syncthreads();
#pragma unroll
    for (int s = 128; s > 0; s >>= 1) {
        if (tid < s) red[tid] += red[tid + s];
        __syncthreads();
    }
    const float inv = 1.0f / red[0];
    __syncthreads();

    // ---- normalize, write p (b128) ----
    float4* pg4 = (float4*)(pout + (size_t)row * D_LAT);
#pragma unroll
    for (int i = tid; i < D_LAT / 4; i += 256) {
        float4 p = *(const float4*)&prow[i * 4];
        p.x *= inv; p.y *= inv; p.z *= inv; p.w *= inv;
        *(float4*)&prow[i * 4] = p;
        pg4[i] = p;
    }
    __syncthreads();

    // ---- 4-level radix select for the CDF-crossing threshold ----
    unsigned prefix_val  = 0u;
    unsigned prefix_mask = 0u;
    float    target      = TAU;
#pragma unroll
    for (int level = 0; level < 4; ++level) {
        const int shift = 24 - 8 * level;
        hist[tid] = 0.0f;
        __syncthreads();
        for (int i = tid; i < D_LAT; i += 256) {
            const float pv = prow[i];
            const unsigned bits = __float_as_uint(pv);
            if ((bits & prefix_mask) == prefix_val)
                atomicAdd(&hist[(bits >> shift) & 255], pv);   // ds_add_f32
        }
        __syncthreads();
        // suffix sums: red[b] = sum of hist[b..255]
        red[tid] = hist[tid];
        __syncthreads();
#pragma unroll
        for (int off = 1; off < 256; off <<= 1) {
            const float v = (tid + off < 256) ? red[tid + off] : 0.0f;
            __syncthreads();
            red[tid] += v;
            __syncthreads();
        }
        if (tid == 0) { schosen = 0; starget = hist[0]; }      // fallback: whole bin 0
        __syncthreads();
        const float Scur  = red[tid];
        const float Snext = (tid < 255) ? red[tid + 1] : 0.0f;
        if (Scur >= target && Snext < target) {                // unique crossing bin
            schosen = tid;
            starget = target - Snext;                          // residual inside this bin
        }
        __syncthreads();
        prefix_val  |= ((unsigned)schosen) << shift;
        prefix_mask |= (255u << shift);
        target = starget;
        __syncthreads();
    }
    const float thr = __uint_as_float(prefix_val);             // k-th largest prob

    // ---- latents = relu(z) * (p >= thr), b128 stores ----
    float4* lg4 = (float4*)(latout + (size_t)row * D_LAT);
#pragma unroll
    for (int i = tid; i < D_LAT / 4; i += 256) {
        const float4 zv = *(const float4*)&zrow[i * 4];
        const float4 p  = *(const float4*)&prow[i * 4];
        float4 o;
        o.x = (p.x >= thr && zv.x > 0.0f) ? zv.x : 0.0f;
        o.y = (p.y >= thr && zv.y > 0.0f) ? zv.y : 0.0f;
        o.z = (p.z >= thr && zv.z > 0.0f) ? zv.z : 0.0f;
        o.w = (p.w >= thr && zv.w > 0.0f) ? zv.w : 0.0f;
        lg4[i] = o;
    }
}

// ===========================================================================
// Kernel 3: recon = tanh(latents @ W_dec + b_dec)
// Block: 256 threads (8 waves). Tile: 64(M) x 128(N), K-step 32, K=4096.
// Wave grid 4(M) x 2(N); each wave owns 1x4 16x16 WMMA tiles.
// ===========================================================================
__global__ __launch_bounds__(256) void sae_gemm2_kernel(
    const float* __restrict__ lat, const float* __restrict__ Wdec,
    const float* __restrict__ bdec, float* __restrict__ recon)
{
    __shared__ __align__(128) __bf16 Alds[64 * 32];    // [m][k]
    __shared__ __align__(128) __bf16 Blds[128 * 32];   // [n][k]

    const int tid  = threadIdx.x;
    const int row0 = blockIdx.x * 64;
    const int wave = tid >> 5;
    const int lane = tid & 31;
    const int wm   = wave & 3;       // M offset wm*16
    const int wn   = wave >> 2;      // N offset wn*64
    const int l16  = lane & 15;
    const int lh   = lane >> 4;

    v8f acc[4];
#pragma unroll
    for (int b = 0; b < 4; ++b) acc[b] = (v8f)(0.0f);

    for (int ks = 0; ks < D_LAT; ks += 32) {
        // ---- stage A: 64x32 fp32 -> bf16 ----
#pragma unroll
        for (int i = 0; i < 2; ++i) {
            const int e  = i * 1024 + tid * 4;
            const int r  = e >> 5;
            const int kk = e & 31;
            const float4 v = *(const float4*)(lat + (size_t)(row0 + r) * D_LAT + ks + kk);
            Alds[r * 32 + kk + 0] = (__bf16)v.x;
            Alds[r * 32 + kk + 1] = (__bf16)v.y;
            Alds[r * 32 + kk + 2] = (__bf16)v.z;
            Alds[r * 32 + kk + 3] = (__bf16)v.w;
        }
        // ---- stage B: 32x128 fp32 -> bf16 transposed [n][k] ----
#pragma unroll
        for (int i = 0; i < 4; ++i) {
            const int e  = i * 1024 + tid * 4;
            const int kr = e >> 7;
            const int nn = e & 127;
            const float4 v = *(const float4*)(Wdec + (size_t)(ks + kr) * D_OUT + nn);
            Blds[(nn + 0) * 32 + kr] = (__bf16)v.x;
            Blds[(nn + 1) * 32 + kr] = (__bf16)v.y;
            Blds[(nn + 2) * 32 + kr] = (__bf16)v.z;
            Blds[(nn + 3) * 32 + kr] = (__bf16)v.w;
        }
        if (ks + 32 < D_LAT)
            __builtin_prefetch(lat + (size_t)(row0 + (tid >> 2)) * D_LAT + ks + 32, 0, 1);
        __syncthreads();

        v16bf afrag;
        {
            const int r  = wm * 16 + l16;
            const int kb = lh * 16;
            const v8bf a0 = *(const v8bf*)(&Alds[r * 32 + kb]);
            const v8bf a1 = *(const v8bf*)(&Alds[r * 32 + kb + 8]);
#pragma unroll
            for (int q = 0; q < 8; ++q) { afrag[q] = a0[q]; afrag[q + 8] = a1[q]; }
        }
#pragma unroll
        for (int tn = 0; tn < 4; ++tn) {
            const int c = wn * 64 + tn * 16 + l16;
            const v16bf bfrag = *(const v16bf*)(&Blds[c * 32 + lh * 16]);
            acc[tn] = __builtin_amdgcn_wmma_f32_16x16x32_bf16(
                false, afrag, false, bfrag, (short)0, acc[tn], false, false);
        }
        __syncthreads();
    }

    // ---- epilogue: tanh(acc + bias) ----
#pragma unroll
    for (int tn = 0; tn < 4; ++tn) {
        const int col = wn * 64 + tn * 16 + l16;
        const float bias = bdec[col];
        const int m0 = row0 + wm * 16 + lh * 8;
#pragma unroll
        for (int r = 0; r < 8; ++r)
            recon[(size_t)(m0 + r) * D_OUT + col] = tanhf(acc[tn][r] + bias);
    }
}

// ===========================================================================
extern "C" void kernel_launch(void* const* d_in, const int* in_sizes, int n_in,
                              void* d_out, int out_size, void* d_ws, size_t ws_size,
                              hipStream_t stream)
{
    (void)in_sizes; (void)n_in; (void)out_size; (void)d_ws; (void)ws_size;

    const float* x    = (const float*)d_in[0];   // (16384, 384)
    const float* Wenc = (const float*)d_in[1];   // (384, 4096)
    const float* benc = (const float*)d_in[2];   // (4096)
    const float* Wdec = (const float*)d_in[3];   // (4096, 128)
    const float* bdec = (const float*)d_in[4];   // (128)

    float* out   = (float*)d_out;
    float* recon = out;                                        // (16384, 128)
    float* p     = out + (size_t)S_ROWS * D_OUT;               // (16384, 4096)
    float* lat   = p   + (size_t)S_ROWS * D_LAT;               // (16384, 4096)

    // z is materialized in the latents region, then overwritten in place.
    sae_gemm1_kernel<<<dim3(S_ROWS / 128, D_LAT / 128), 256, 0, stream>>>(x, Wenc, benc, lat);
    sae_topk_kernel<<<dim3(S_ROWS), 256, 0, stream>>>(lat, p, lat);
    sae_gemm2_kernel<<<dim3(S_ROWS / 64), 256, 0, stream>>>(lat, Wdec, bdec, recon);
}
